// GraphChannelEmbed_249108103808
// MI455X (gfx1250) — matrix operand order repaired and checked by reading, not compiled, not run on hardware
//
#include <hip/hip_runtime.h>
#include <hip/hip_bf16.h>
#include <math.h>

// ---------------------------------------------------------------------------
// Problem constants (reference: B=8, H=W=64, C=512)
// ---------------------------------------------------------------------------
#define BATCH 8
#define GH 64
#define GW 64
#define NNODE (GH * GW)       // 4096
#define CH 512
#define MROWS (BATCH * NNODE) // 32768

// GEMM tiling
#define BM 128
#define BN 128
#define BK 32

typedef __bf16 bf16;
typedef bf16  v16bf __attribute__((ext_vector_type(16)));
typedef bf16  v8bf  __attribute__((ext_vector_type(8)));
typedef float v8f   __attribute__((ext_vector_type(8)));

__device__ __forceinline__ v8f zero8() {
    v8f z;
#pragma unroll
    for (int i = 0; i < 8; ++i) z[i] = 0.0f;
    return z;
}

// ---------------------------------------------------------------------------
// K1: C[M,N] = A[M,K] (fp32) @ W[K,N] (fp32) (+ bias), bf16 WMMA inner product.
// Block tile 128x128, 256 threads = 8 waves (2x4), wave tile 64x32 (4x2 WMMA).
// Register-staged pipeline: global loads for tile k+1 issue before the WMMAs
// of tile k; conversion + LDS store happen after the compute barrier, so the
// loadcnt wait is covered by 8 WMMAs of useful work.
// ---------------------------------------------------------------------------
__global__ __launch_bounds__(256)
void gemm_bf16_wmma(const float* __restrict__ A, const float* __restrict__ W,
                    const float* __restrict__ bias, float* __restrict__ C,
                    int M, int K, int N) {
    __shared__ __align__(16) bf16 ldsA[BM * BK]; // [m][k]
    __shared__ __align__(16) bf16 ldsB[BN * BK]; // [n][k] (transposed)

    const int tid  = threadIdx.x;
    const int wave = tid >> 5;
    const int lane = tid & 31;
    const int bm = blockIdx.x * BM;
    const int bn = blockIdx.y * BN;
    const int wm = (wave >> 2) * 64; // 0 / 64
    const int wn = (wave & 3) * 32;  // 0..96

    // staging-register mapping
    const int sa_r  = tid >> 3;       // A: rows sa_r + 32*i
    const int sa_c4 = (tid & 7) * 4;  // A: k-offset within tile
    const int sb_n  = tid & 127;      // B: column within tile
    const int sb_kq = (tid >> 7);     // B: k-quad base (0/1), quads kq+2*i

    float4 ra[4];
    float  rb[16];

    auto loadTile = [&](int kb) {
#pragma unroll
        for (int i = 0; i < 4; ++i)
            ra[i] = *(const float4*)(A + (size_t)(bm + sa_r + 32 * i) * K + kb + sa_c4);
#pragma unroll
        for (int i = 0; i < 4; ++i) {
            const int k0 = (sb_kq + 2 * i) * 4;
#pragma unroll
            for (int q = 0; q < 4; ++q)
                rb[4 * i + q] = W[(size_t)(kb + k0 + q) * N + bn + sb_n];
        }
    };
    auto storeTile = [&]() {
#pragma unroll
        for (int i = 0; i < 4; ++i) {
            bf16* d = &ldsA[(sa_r + 32 * i) * BK + sa_c4];
            d[0] = (bf16)ra[i].x; d[1] = (bf16)ra[i].y;
            d[2] = (bf16)ra[i].z; d[3] = (bf16)ra[i].w;
        }
#pragma unroll
        for (int i = 0; i < 4; ++i) {
            const int k0 = (sb_kq + 2 * i) * 4;
            bf16* d = &ldsB[sb_n * BK + k0];
            d[0] = (bf16)rb[4 * i];     d[1] = (bf16)rb[4 * i + 1];
            d[2] = (bf16)rb[4 * i + 2]; d[3] = (bf16)rb[4 * i + 3];
        }
    };

    v8f acc[4][2];
#pragma unroll
    for (int i = 0; i < 4; ++i)
#pragma unroll
        for (int j = 0; j < 2; ++j) acc[i][j] = zero8();

    const int lm = lane & 15;
    const int lk = lane >> 4;

    // prologue: stage tile 0
    loadTile(0);
    storeTile();
    __syncthreads();

    for (int kb = 0; kb < K; kb += BK) {
        const bool more = (kb + BK) < K;
        if (more) loadTile(kb + BK); // in-flight during the WMMAs below

        // ---- fragments ----
        // A 16x32 bf16: lane<16 -> row m, elems {K0..7, K16..23}; lane>=16 -> {K8..15, K24..31}
        v16bf afrag[4];
#pragma unroll
        for (int i = 0; i < 4; ++i) {
            const bf16* base = &ldsA[(wm + i * 16 + lm) * BK];
            v8bf lo = *(const v8bf*)(base + lk * 8);
            v8bf hi = *(const v8bf*)(base + 16 + lk * 8);
            afrag[i] = __builtin_shufflevector(lo, hi, 0, 1, 2, 3, 4, 5, 6, 7,
                                               8, 9, 10, 11, 12, 13, 14, 15);
        }
        // B 32x16 bf16: lane<16 -> col n, elems {K0..7, K16..23}; lane>=16 -> {K8..15, K24..31}
        v16bf bfrag[2];
#pragma unroll
        for (int j = 0; j < 2; ++j) {
            const bf16* base = &ldsB[(wn + j * 16 + lm) * BK];
            v8bf lo = *(const v8bf*)(base + lk * 8);
            v8bf hi = *(const v8bf*)(base + 16 + lk * 8);
            bfrag[j] = __builtin_shufflevector(lo, hi, 0, 1, 2, 3, 4, 5, 6, 7,
                                               8, 9, 10, 11, 12, 13, 14, 15);
        }

#pragma unroll
        for (int i = 0; i < 4; ++i)
#pragma unroll
            for (int j = 0; j < 2; ++j)
                acc[i][j] = __builtin_amdgcn_wmma_f32_16x16x32_bf16(
                    false, afrag[i], false, bfrag[j], (short)0, acc[i][j], false, false);

        __syncthreads();           // all LDS reads of this tile done
        if (more) storeTile();     // waits loadcnt here, cvt_pk + packed ds stores
        __syncthreads();
    }

    // ---- epilogue: D layout VGPR e -> M = e + (lane>=16 ? 8:0), N = lane&15 ----
    const int lhi = lk * 8;
#pragma unroll
    for (int j = 0; j < 2; ++j) {
        const int col = bn + wn + j * 16 + lm;
        const float bv = bias ? bias[col] : 0.0f;
#pragma unroll
        for (int i = 0; i < 4; ++i) {
#pragma unroll
            for (int e = 0; e < 8; ++e) {
                const int row = bm + wm + i * 16 + lhi + e;
                C[(size_t)row * N + col] = acc[i][j][e] + bv;
            }
        }
    }
}

// ---------------------------------------------------------------------------
// K2: GCN aggregation on the 64x64 grid. message = dinv[src]*dinv[dst]*xw[src];
// mean over incoming edges (4-neighbors + self), + bias, ReLU.
// One block per (node, batch); 256 threads cover 512 channels. The whole xw
// tensor (64 MB) is L2-resident, so neighbor re-reads stay on-chip.
// ---------------------------------------------------------------------------
__global__ __launch_bounds__(256)
void gcn_aggregate(const float* __restrict__ xw, const float* __restrict__ bias,
                   float* __restrict__ out) {
    const int n = blockIdx.x;   // 0..4095
    const int b = blockIdx.y;   // 0..7
    const int i = n >> 6, j = n & 63;

    auto dinv = [](int ii, int jj) -> float {
        float d = 1.0f + (ii > 0) + (ii < GH - 1) + (jj > 0) + (jj < GW - 1);
        return rsqrtf(d);
    };
    const float ds   = dinv(i, j);
    const float deg  = 1.0f + (i > 0) + (i < GH - 1) + (j > 0) + (j < GW - 1);
    const float wUp = (i > 0)      ? dinv(i - 1, j) : 0.0f;
    const float wDn = (i < GH - 1) ? dinv(i + 1, j) : 0.0f;
    const float wLf = (j > 0)      ? dinv(i, j - 1) : 0.0f;
    const float wRt = (j < GW - 1) ? dinv(i, j + 1) : 0.0f;

    const float* base = xw + (size_t)b * NNODE * CH;
    const float scale = ds / deg;

    for (int c = threadIdx.x; c < CH; c += 256) {
        float s = ds * base[(size_t)n * CH + c];
        if (i > 0)      s += wUp * base[(size_t)(n - GW) * CH + c];
        if (i < GH - 1) s += wDn * base[(size_t)(n + GW) * CH + c];
        if (j > 0)      s += wLf * base[(size_t)(n - 1) * CH + c];
        if (j < GW - 1) s += wRt * base[(size_t)(n + 1) * CH + c];
        out[(size_t)b * NNODE * CH + (size_t)n * CH + c] =
            fmaxf(0.0f, s * scale + bias[c]);
    }
}

// ---------------------------------------------------------------------------
// K0: zero the small accumulators (ws is poisoned by the harness)
// ---------------------------------------------------------------------------
__global__ void zero_small(float* p, int n) {
    int i = blockIdx.x * blockDim.x + threadIdx.x;
    if (i < n) p[i] = 0.0f;
}

// K3a: pooled[b][c] += sum over an n-slice (atomic partials, coalesced over c)
__global__ __launch_bounds__(256)
void pool_accum(const float* __restrict__ h, float* __restrict__ pooled) {
    const int c = blockIdx.x * 256 + threadIdx.x;
    const int b = blockIdx.z;
    const int n0 = blockIdx.y * 256;
    float s = 0.0f;
    for (int n = n0; n < n0 + 256; ++n)
        s += h[((size_t)b * NNODE + n) * CH + c];
    atomicAdd(&pooled[b * CH + c], s);
}

// K3b: BN over batch of pooled means -> xg[b][c]
__global__ void bn1_kernel(const float* __restrict__ pooled,
                           const float* __restrict__ g1,
                           const float* __restrict__ beta1,
                           float* __restrict__ xg) {
    const int c = threadIdx.x; // 512 threads
    float p[BATCH], mu = 0.0f;
#pragma unroll
    for (int b = 0; b < BATCH; ++b) {
        p[b] = pooled[b * CH + c] * (1.0f / NNODE);
        mu += p[b];
    }
    mu *= (1.0f / BATCH);
    float var = 0.0f;
#pragma unroll
    for (int b = 0; b < BATCH; ++b) { float d = p[b] - mu; var += d * d; }
    var *= (1.0f / BATCH);
    const float inv = rsqrtf(var + 1e-5f);
#pragma unroll
    for (int b = 0; b < BATCH; ++b)
        xg[b * CH + c] = (p[b] - mu) * inv * g1[c] + beta1[c];
}

// K4a: per-channel sum / sumsq of y = residual[b,n,c] + xg[b,c]
__global__ __launch_bounds__(256)
void bn2_stats(const float* __restrict__ residual, const float* __restrict__ xg,
               float* __restrict__ sum, float* __restrict__ sumsq) {
    const int c = blockIdx.x * 256 + threadIdx.x;
    const int b = blockIdx.z;
    const int n0 = blockIdx.y * 256;
    const float xgv = xg[b * CH + c];
    float s = 0.0f, s2 = 0.0f;
    for (int n = n0; n < n0 + 256; ++n) {
        float y = residual[((size_t)b * NNODE + n) * CH + c] + xgv;
        s += y; s2 += y * y;
    }
    atomicAdd(&sum[c], s);
    atomicAdd(&sumsq[c], s2);
}

// K4b: fold stats into per-channel affine: y*scale + shift
__global__ void bn2_params(const float* __restrict__ sum, const float* __restrict__ sumsq,
                           const float* __restrict__ g2, const float* __restrict__ beta2,
                           float* __restrict__ params) {
    const int c = threadIdx.x; // 512 threads
    const float inv_n = 1.0f / (float)(BATCH * NNODE);
    const float mean = sum[c] * inv_n;
    const float var = fmaxf(0.0f, sumsq[c] * inv_n - mean * mean);
    const float sc = g2[c] * rsqrtf(var + 1e-5f);
    params[2 * c]     = sc;
    params[2 * c + 1] = beta2[c] - mean * sc;
}

// K5: [B,N,C] -> [B,C,H,W] transpose through LDS + BN2 affine
__global__ __launch_bounds__(256)
void bn2_apply(const float* __restrict__ residual, const float* __restrict__ xg,
               const float* __restrict__ params, float* __restrict__ out) {
    __shared__ float tile[32][33];
    const int b  = blockIdx.z;
    const int n0 = blockIdx.x * 32;
    const int c0 = blockIdx.y * 32;
    const int tid = threadIdx.x;

#pragma unroll
    for (int it = 0; it < 4; ++it) { // read: consecutive threads -> consecutive c
        const int l  = tid + 256 * it;
        const int nn = l >> 5, cc = l & 31;
        tile[nn][cc] = residual[((size_t)b * NNODE + n0 + nn) * CH + c0 + cc];
    }
    __syncthreads();
#pragma unroll
    for (int it = 0; it < 4; ++it) { // write: consecutive threads -> consecutive n
        const int l  = tid + 256 * it;
        const int cc = l >> 5, nn = l & 31;
        const int c = c0 + cc;
        const float y = tile[nn][cc] + xg[b * CH + c];
        out[((size_t)b * CH + c) * NNODE + n0 + nn] = y * params[2 * c] + params[2 * c + 1];
    }
}

// ---------------------------------------------------------------------------
// Launch sequence
// ---------------------------------------------------------------------------
extern "C" void kernel_launch(void* const* d_in, const int* in_sizes, int n_in,
                              void* d_out, int out_size, void* d_ws, size_t ws_size,
                              hipStream_t stream) {
    const float* x     = (const float*)d_in[0];
    const float* w_res = (const float*)d_in[3];
    const float* w_pre = (const float*)d_in[4];
    const float* b_pre = (const float*)d_in[5];
    const float* w_g1  = (const float*)d_in[6];
    const float* b_g1  = (const float*)d_in[7];
    const float* w_g2  = (const float*)d_in[8];
    const float* b_g2  = (const float*)d_in[9];
    const float* g1    = (const float*)d_in[10];
    const float* beta1 = (const float*)d_in[11];
    const float* g2    = (const float*)d_in[12];
    const float* beta2 = (const float*)d_in[13];

    float* ws = (float*)d_ws;
    const size_t big = (size_t)MROWS * CH; // 16.7M floats
    float* res    = ws;                    // residual (kept to the end)
    float* bufA   = ws + big;              // xr / h1 / h2
    float* bufB   = ws + 2 * big;          // xw scratch
    float* pooled = ws + 3 * big;          // [8][512]
    float* xg     = pooled + BATCH * CH;   // [8][512]
    float* sum    = xg + BATCH * CH;       // [512]
    float* sumsq  = sum + CH;              // [512]
    float* params = sumsq + CH;            // [512][2]

    // zero 8*512+2*512 floats from pooled: pooled + first 1024 of xg (sum/sumsq lie past it)
    zero_small<<<(BATCH * CH + 2 * CH + 255) / 256, 256, 0, stream>>>(pooled,
                                                                     BATCH * CH + 2 * CH);

    const dim3 gg(MROWS / BM, CH / BN);          // 256 x 4 blocks
    const dim3 ga(NNODE, BATCH);                 // stencil grid
    const dim3 gr(CH / 256, NNODE / 256, BATCH); // reduction grid (2,16,8)

    // residual = x @ w_res ; xr = x @ w_pre + b_pre
    gemm_bf16_wmma<<<gg, 256, 0, stream>>>(x, w_res, nullptr, res,  MROWS, CH, CH);
    gemm_bf16_wmma<<<gg, 256, 0, stream>>>(x, w_pre, b_pre,  bufA, MROWS, CH, CH);

    // GCN layer 1: xw = xr @ w_g1 ; h1 = relu(agg(xw) + b_g1)
    gemm_bf16_wmma<<<gg, 256, 0, stream>>>(bufA, w_g1, nullptr, bufB, MROWS, CH, CH);
    gcn_aggregate<<<ga, 256, 0, stream>>>(bufB, b_g1, bufA);

    // GCN layer 2: xw = h1 @ w_g2 ; h2 = relu(agg(xw) + b_g2)
    gemm_bf16_wmma<<<gg, 256, 0, stream>>>(bufA, w_g2, nullptr, bufB, MROWS, CH, CH);
    gcn_aggregate<<<ga, 256, 0, stream>>>(bufB, b_g2, bufA);

    // pool + BN over batch -> xg
    pool_accum<<<gr, 256, 0, stream>>>(bufA, pooled);
    bn1_kernel<<<1, CH, 0, stream>>>(pooled, g1, beta1, xg);

    // BN2 over (B,H,W) of residual + xg, then transpose-store [B,C,H,W]
    bn2_stats<<<gr, 256, 0, stream>>>(res, xg, sum, sumsq);
    bn2_params<<<1, CH, 0, stream>>>(sum, sumsq, g2, beta2, params);
    bn2_apply<<<dim3(NNODE / 32, CH / 32, BATCH), 256, 0, stream>>>(res, xg, params,
                                                                    (float*)d_out);
}
